// AttentionBlock_37340445672198
// MI455X (gfx1250) — matrix operand order repaired and checked by reading, not compiled, not run on hardware
//
#include <hip/hip_runtime.h>
#include <hip/hip_bf16.h>

typedef _Float16 h16;
typedef __attribute__((ext_vector_type(16))) _Float16 v16h;
typedef __attribute__((ext_vector_type(8)))  _Float16 v8h;
typedef __attribute__((ext_vector_type(8)))  float    v8f;

#define WMMA_F16(A_, B_, C_) \
  __builtin_amdgcn_wmma_f32_16x16x32_f16(false, (A_), false, (B_), (short)0, (C_), false, false)

static constexpr int Bn = 2, Tn = 2048, Dn = 1024, Hn = 16, DHn = 64;

// ---------------- fragment loaders (ISA 7.12.2 layouts) ----------------
// A: 16x32 f16, row-major source, lda in elements.
__device__ __forceinline__ v16h load_a_frag(const h16* base, int lda, int mBase, int kBase, int lane) {
  int m = lane & 15;
  int h8 = (lane >> 4) * 8;
  const h16* p = base + (size_t)(mBase + m) * lda + kBase + h8;
  v8h lo = *(const v8h*)p;
  v8h hi = *(const v8h*)(p + 16);
  return __builtin_shufflevector(lo, hi, 0,1,2,3,4,5,6,7,8,9,10,11,12,13,14,15);
}
// B: 32x16 f16 from K-contiguous ("transposed", [N,K]) source, ldb in elements.
__device__ __forceinline__ v16h load_b_fragT(const h16* bt, int ldb, int nBase, int kBase, int lane) {
  int n = lane & 15;
  int kh = (lane >> 4) * 8;
  const h16* p = bt + (size_t)(nBase + n) * ldb + kBase + kh;
  v8h lo = *(const v8h*)p;
  v8h hi = *(const v8h*)(p + 16);
  return __builtin_shufflevector(lo, hi, 0,1,2,3,4,5,6,7,8,9,10,11,12,13,14,15);
}
// B fragment from a [64 rows][32 K] f16 LDS stage (row = N, K contiguous).
__device__ __forceinline__ v16h load_b_frag_lds(const h16* stage, int nSub, int lane) {
  int n = lane & 15;
  int kh = (lane >> 4) * 8;
  const h16* p = stage + (nSub * 16 + n) * 32 + kh;
  v8h lo = *(const v8h*)p;
  v8h hi = *(const v8h*)(p + 16);
  return __builtin_shufflevector(lo, hi, 0,1,2,3,4,5,6,7,8,9,10,11,12,13,14,15);
}

// ---------------- async global->LDS staging of a 64Nx32K f16 B tile ----------------
// 64 rows * 64B/row = 256 x B128 transfers = 8 per lane. ASYNCcnt tracked.
__device__ __forceinline__ void async_stage_B(const h16* Bt, int K, int nBase, int k0,
                                              uint32_t lds_base, int lane) {
#pragma unroll
  for (int t = 0; t < 8; t++) {
    int id = t * 32 + lane;
    int r = id >> 2, c = id & 3;                       // row 0..63, 16B chunk 0..3
    uint64_t g = (uint64_t)(uintptr_t)(Bt + (size_t)(nBase + r) * K + k0 + c * 8);
    uint32_t l = lds_base + (uint32_t)(r * 64 + c * 16);
    asm volatile("global_load_async_to_lds_b128 %0, %1, off" :: "v"(l), "v"(g) : "memory");
  }
}

// ---------------- shared GEMM core: 32M x 64N wave tile, async-staged B ----------------
// acc[0..3] = rows [mBase..mBase+15], acc[4..7] = rows [mBase+16..mBase+31]
__device__ __forceinline__ void gemm_core(const h16* __restrict__ A, const h16* __restrict__ Bt,
                                          int K, int mBase, int nBase, int lane,
                                          h16* stage0, h16* stage1, v8f acc[8]) {
  uint32_t sb0 = (uint32_t)(uintptr_t)stage0;
  uint32_t sb1 = (uint32_t)(uintptr_t)stage1;
  async_stage_B(Bt, K, nBase, 0, sb0, lane);
  for (int k0 = 0, step = 0; k0 < K; k0 += 32, step++) {
    int buf = step & 1;
    h16* st = buf ? stage1 : stage0;
    if (k0 + 32 < K) {
      async_stage_B(Bt, K, nBase, k0 + 32, buf ? sb0 : sb1, lane);
      asm volatile("s_wait_asynccnt 0x8" ::: "memory");   // current tile landed (in-order)
    } else {
      asm volatile("s_wait_asynccnt 0x0" ::: "memory");
    }
    v16h a0 = load_a_frag(A, K, mBase, k0, lane);
    v16h a1 = load_a_frag(A, K, mBase + 16, k0, lane);
#pragma unroll
    for (int j = 0; j < 4; j++) {
      v16h bf = load_b_frag_lds(st, j, lane);
      acc[j]     = WMMA_F16(a0, bf, acc[j]);
      acc[4 + j] = WMMA_F16(a1, bf, acc[4 + j]);
    }
  }
}

// ---------------- weight cast kernels ----------------
__global__ __launch_bounds__(256) void cast_T(const float* __restrict__ src, h16* __restrict__ dst,
                                              int rows, int cols) {
  int idx = blockIdx.x * 256 + threadIdx.x;
  int r = idx / cols, c = idx % cols;
  dst[(size_t)c * rows + r] = (h16)src[idx];
}
__global__ __launch_bounds__(256) void cast_qkvw(const float* __restrict__ src, h16* __restrict__ dst) {
  int idx = blockIdx.x * 256 + threadIdx.x;      // H*D*DH = 1M
  int h = idx >> 16;                             // D*DH = 65536
  int rem = idx & 65535;
  int d = rem >> 6, e = rem & 63;
  dst[((size_t)(h * DHn + e)) * Dn + d] = (h16)src[idx];
}

// ---------------- LayerNorm (optionally + f16 residual) -> f16 ----------------
__global__ __launch_bounds__(256) void ln_kernel(const float* __restrict__ x,
                                                 const h16* __restrict__ res,
                                                 const float* __restrict__ g,
                                                 const float* __restrict__ bb,
                                                 h16* __restrict__ out) {
  int lane = threadIdx.x & 31, wave = threadIdx.x >> 5;
  int row = blockIdx.x * 8 + wave;
  const float* xr = x + (size_t)row * Dn;
  float vals[32];
  float s = 0.f, s2 = 0.f;
#pragma unroll
  for (int i = 0; i < 8; i++) {
    int idx = i * 128 + lane * 4;
    float4 xv = *(const float4*)(xr + idx);
    float v0 = xv.x, v1 = xv.y, v2 = xv.z, v3 = xv.w;
    if (res) {
      const h16* rr = res + (size_t)row * Dn + idx;
      v0 += (float)rr[0]; v1 += (float)rr[1]; v2 += (float)rr[2]; v3 += (float)rr[3];
    }
    vals[i*4+0] = v0; vals[i*4+1] = v1; vals[i*4+2] = v2; vals[i*4+3] = v3;
    s  += v0 + v1 + v2 + v3;
    s2 += v0*v0 + v1*v1 + v2*v2 + v3*v3;
  }
#pragma unroll
  for (int off = 1; off < 32; off <<= 1) { s += __shfl_xor(s, off, 32); s2 += __shfl_xor(s2, off, 32); }
  float mu = s * (1.f / Dn);
  float var = s2 * (1.f / Dn) - mu * mu;
  float rs = rsqrtf(var + 1e-5f);
#pragma unroll
  for (int i = 0; i < 8; i++) {
    int idx = i * 128 + lane * 4;
#pragma unroll
    for (int j = 0; j < 4; j++)
      out[(size_t)row * Dn + idx + j] = (h16)((vals[i*4+j] - mu) * rs * g[idx + j] + bb[idx + j]);
  }
}

// ---------------- fused QKV GEMM: [4096,1024] x [1024,3072] ----------------
__global__ __launch_bounds__(256) void gemm_qkv(const h16* __restrict__ A, const h16* __restrict__ Bt,
                                                h16* __restrict__ q16, h16* __restrict__ k16,
                                                h16* __restrict__ v16t) {
  __shared__ h16 Bstage[8][2][64 * 32];
  int lane = threadIdx.x & 31, wave = threadIdx.x >> 5;
  int mBase = blockIdx.x * 128 + (wave & 3) * 32;
  int nBase = blockIdx.y * 128 + (wave >> 2) * 64;
  v8f acc[8] = {};
  gemm_core(A, Bt, Dn, mBase, nBase, lane, &Bstage[wave][0][0], &Bstage[wave][1][0], acc);
  int n = lane & 15, m0 = (lane >> 4) * 8;
#pragma unroll
  for (int ms = 0; ms < 2; ms++)
#pragma unroll
    for (int j = 0; j < 4; j++)
#pragma unroll
      for (int i = 0; i < 8; i++) {
        int ng = nBase + j * 16 + n;
        int row = mBase + ms * 16 + m0 + i;
        float val = acc[ms * 4 + j][i];
        int sect = ng >> 10, c = ng & 1023;
        int h = c >> 6, e = c & 63;
        int b = row >> 11, t = row & 2047;
        size_t bh = (size_t)(b * Hn + h);
        if (sect == 0)      q16[(bh * Tn + t) * DHn + e] = (h16)(val * 0.125f);  // fold DH^-0.5
        else if (sect == 1) k16[(bh * Tn + t) * DHn + e] = (h16)val;
        else                v16t[(bh * DHn + e) * Tn + t] = (h16)val;            // V transposed
      }
}

// ---------------- flash attention: one wave = 16 query rows ----------------
__global__ __launch_bounds__(128) void flash_attn(const h16* __restrict__ q16,
                                                  const h16* __restrict__ k16,
                                                  const h16* __restrict__ v16t,
                                                  h16* __restrict__ attn16) {
  __shared__ _Float16 Pb[4][16][32];
  int lane = threadIdx.x & 31, wave = threadIdx.x >> 5;
  int bh = blockIdx.x;
  int qBase = blockIdx.y * 64 + wave * 16;
  const h16* qp = q16 + (size_t)bh * Tn * DHn;
  const h16* kp = k16 + (size_t)bh * Tn * DHn;
  const h16* vp = v16t + (size_t)bh * DHn * Tn;
  int n = lane & 15, hl = lane >> 4, m0 = hl * 8;

  v16h aQ0 = load_a_frag(qp, DHn, qBase, 0, lane);
  v16h aQ1 = load_a_frag(qp, DHn, qBase, 32, lane);
  v8f acc[4] = {};
  float mprev[8], lsum[8];
#pragma unroll
  for (int i = 0; i < 8; i++) { mprev[i] = -1e30f; lsum[i] = 0.f; }

  for (int kb = 0; kb <= qBase + 15; kb += 32) {
    v8f S0 = {}, S1 = {};
    {
      v16h b0 = load_b_fragT(kp, DHn, kb, 0, lane);
      v16h b1 = load_b_fragT(kp, DHn, kb, 32, lane);
      S0 = WMMA_F16(aQ0, b0, S0);
      S0 = WMMA_F16(aQ1, b1, S0);
    }
    {
      v16h b0 = load_b_fragT(kp, DHn, kb + 16, 0, lane);
      v16h b1 = load_b_fragT(kp, DHn, kb + 16, 32, lane);
      S1 = WMMA_F16(aQ0, b0, S1);
      S1 = WMMA_F16(aQ1, b1, S1);
    }
    if (kb + 31 > qBase) {   // causal mask (uniform branch per wave)
#pragma unroll
      for (int i = 0; i < 8; i++) {
        int qi = qBase + m0 + i;
        if (kb + n > qi)      S0[i] = -1e30f;
        if (kb + 16 + n > qi) S1[i] = -1e30f;
      }
    }
    float sc[8];
#pragma unroll
    for (int i = 0; i < 8; i++) {
      float rmax = fmaxf(S0[i], S1[i]);
#pragma unroll
      for (int off = 1; off < 16; off <<= 1) rmax = fmaxf(rmax, __shfl_xor(rmax, off, 32));
      float mnew = fmaxf(mprev[i], rmax);
      float p0 = __expf(S0[i] - mnew), p1 = __expf(S1[i] - mnew);
      float rs = p0 + p1;
#pragma unroll
      for (int off = 1; off < 16; off <<= 1) rs += __shfl_xor(rs, off, 32);
      sc[i] = __expf(mprev[i] - mnew);
      lsum[i] = lsum[i] * sc[i] + rs;
      mprev[i] = mnew;
      Pb[wave][m0 + i][n]      = (h16)p0;   // C-layout -> LDS (row-major P tile)
      Pb[wave][m0 + i][16 + n] = (h16)p1;
    }
#pragma unroll
    for (int j = 0; j < 4; j++)
#pragma unroll
      for (int i = 0; i < 8; i++) acc[j][i] *= sc[i];

    asm volatile("s_wait_dscnt 0" ::: "memory");  // LDS P writes visible before A-frag reads

    const _Float16* pp = &Pb[wave][lane & 15][hl * 8];
    v8h plo = *(const v8h*)pp;
    v8h phi = *(const v8h*)(pp + 16);
    v16h aP = __builtin_shufflevector(plo, phi, 0,1,2,3,4,5,6,7,8,9,10,11,12,13,14,15);
#pragma unroll
    for (int j = 0; j < 4; j++) {
      v16h bV = load_b_fragT(vp, Tn, j * 16, kb, lane);   // [DH,T]: contiguous along keys
      acc[j] = WMMA_F16(aP, bV, acc[j]);
    }
  }

  int b = bh >> 4, h = bh & 15;
#pragma unroll
  for (int i = 0; i < 8; i++) {
    float inv = 1.f / lsum[i];
    int t = qBase + m0 + i;
    size_t base = ((size_t)b * Tn + t) * Dn + h * DHn;
#pragma unroll
    for (int j = 0; j < 4; j++) attn16[base + j * 16 + n] = (h16)(acc[j][i] * inv);
  }
}

// ---------------- FFN GEMMs ----------------
__global__ __launch_bounds__(256) void gemm_bias_gelu(const h16* __restrict__ A, const h16* __restrict__ Bt,
                                                      const float* __restrict__ bias, h16* __restrict__ out,
                                                      int K, int N) {
  __shared__ h16 Bstage[8][2][64 * 32];
  int lane = threadIdx.x & 31, wave = threadIdx.x >> 5;
  int mBase = blockIdx.x * 128 + (wave & 3) * 32;
  int nBase = blockIdx.y * 128 + (wave >> 2) * 64;
  v8f acc[8] = {};
  gemm_core(A, Bt, K, mBase, nBase, lane, &Bstage[wave][0][0], &Bstage[wave][1][0], acc);
  int n = lane & 15, m0 = (lane >> 4) * 8;
#pragma unroll
  for (int ms = 0; ms < 2; ms++)
#pragma unroll
    for (int j = 0; j < 4; j++)
#pragma unroll
      for (int i = 0; i < 8; i++) {
        int ng = nBase + j * 16 + n;
        int row = mBase + ms * 16 + m0 + i;
        float y = acc[ms * 4 + j][i] + bias[ng];
        y = 0.5f * y * (1.0f + erff(y * 0.70710678118654752f));  // exact GELU
        out[(size_t)row * N + ng] = (h16)y;
      }
}

__global__ __launch_bounds__(256) void gemm_bias_res(const h16* __restrict__ A, const h16* __restrict__ Bt,
                                                     const float* __restrict__ bias,
                                                     const float* __restrict__ resx, float* __restrict__ out,
                                                     int K, int N) {
  __shared__ h16 Bstage[8][2][64 * 32];
  int lane = threadIdx.x & 31, wave = threadIdx.x >> 5;
  int mBase = blockIdx.x * 128 + (wave & 3) * 32;
  int nBase = blockIdx.y * 128 + (wave >> 2) * 64;
  v8f acc[8] = {};
  gemm_core(A, Bt, K, mBase, nBase, lane, &Bstage[wave][0][0], &Bstage[wave][1][0], acc);
  int n = lane & 15, m0 = (lane >> 4) * 8;
#pragma unroll
  for (int ms = 0; ms < 2; ms++)
#pragma unroll
    for (int j = 0; j < 4; j++)
#pragma unroll
      for (int i = 0; i < 8; i++) {
        int ng = nBase + j * 16 + n;
        int row = mBase + ms * 16 + m0 + i;
        size_t o = (size_t)row * N + ng;
        out[o] = resx[o] + acc[ms * 4 + j][i] + bias[ng];   // residual from ORIGINAL x
      }
}

// ---------------- launch ----------------
extern "C" void kernel_launch(void* const* d_in, const int* in_sizes, int n_in,
                              void* d_out, int out_size, void* d_ws, size_t ws_size,
                              hipStream_t stream) {
  (void)in_sizes; (void)n_in; (void)out_size; (void)ws_size;
  const float* x    = (const float*)d_in[0];
  // d_in[1] = causal mask (bool) — implicit in flash_attn
  const float* ln1g = (const float*)d_in[2];
  const float* ln1b = (const float*)d_in[3];
  const float* ln2g = (const float*)d_in[4];
  const float* ln2b = (const float*)d_in[5];
  const float* Wq   = (const float*)d_in[6];
  const float* Wk   = (const float*)d_in[7];
  const float* Wv   = (const float*)d_in[8];
  const float* W1   = (const float*)d_in[9];
  const float* b1   = (const float*)d_in[10];
  const float* W2   = (const float*)d_in[11];
  const float* b2   = (const float*)d_in[12];
  float* out = (float*)d_out;

  char* ws = (char*)d_ws;
  const size_t MB = (size_t)1 << 20;
  // Region 0..38MB reused: {xnorm,wqkvt,q16,k16,v16t} live only until flash_attn,
  // then mid16 (32MB) overlays it.
  h16* xnorm  = (h16*)(ws + 0 * MB);   // 8 MB  [4096,1024]
  h16* wqkvt  = (h16*)(ws + 8 * MB);   // 6 MB  [3072,1024]
  h16* q16    = (h16*)(ws + 14 * MB);  // 8 MB  [B,H,T,DH]
  h16* k16    = (h16*)(ws + 22 * MB);  // 8 MB  [B,H,T,DH]
  h16* v16t   = (h16*)(ws + 30 * MB);  // 8 MB  [B,H,DH,T]
  h16* mid16  = (h16*)(ws + 0 * MB);   // 32 MB [4096,4096] (overlays the above)
  h16* attn16 = (h16*)(ws + 38 * MB);  // 8 MB  [B,T,D]
  h16* h1     = (h16*)(ws + 46 * MB);  // 8 MB  [4096,1024]
  h16* w1t    = (h16*)(ws + 54 * MB);  // 8 MB  [4096,1024]
  h16* w2t    = (h16*)(ws + 62 * MB);  // 8 MB  [1024,4096]  -> 70 MB total

  cast_qkvw<<<4096, 256, 0, stream>>>(Wq, wqkvt);
  cast_qkvw<<<4096, 256, 0, stream>>>(Wk, wqkvt + (size_t)Dn * Dn);
  cast_qkvw<<<4096, 256, 0, stream>>>(Wv, wqkvt + (size_t)2 * Dn * Dn);
  cast_T<<<16384, 256, 0, stream>>>(W1, w1t, Dn, 4 * Dn);
  cast_T<<<16384, 256, 0, stream>>>(W2, w2t, 4 * Dn, Dn);

  ln_kernel<<<512, 256, 0, stream>>>(x, nullptr, ln1g, ln1b, xnorm);
  gemm_qkv<<<dim3(32, 24), 256, 0, stream>>>(xnorm, wqkvt, q16, k16, v16t);
  flash_attn<<<dim3(Bn * Hn, Tn / 64), 128, 0, stream>>>(q16, k16, v16t, attn16);
  ln_kernel<<<512, 256, 0, stream>>>(x, attn16, ln2g, ln2b, h1);
  gemm_bias_gelu<<<dim3(32, 32), 256, 0, stream>>>(h1, w1t, b1, mid16, Dn, 4 * Dn);
  gemm_bias_res<<<dim3(32, 8), 256, 0, stream>>>(mid16, w2t, b2, x, out, 4 * Dn, Dn);
}
